// MultiHeadAttention_59339268161893
// MI455X (gfx1250) — compile-verified, matrix operand order checked by reading, not run on hardware
//
#include <hip/hip_runtime.h>

// MI455X / gfx1250 multi-head attention forward.
// GEMMs: v_wmma_f32_16x16x32_f16, register-blocked 16x64 per wave, with the
// weight k-slab staged into double-buffered LDS via global_load_async_to_lds
// (ASYNCcnt) — builtin confirmed present on this toolchain.
// Workspace (f16): x_q,x_k,x_v (4096x768), Wq,Wk,Wv,Wo (768x768),
// Qh,Kh [b][h][s][64], Vt [b][h][64][s], attn-out (4096x768)  => ~48.8 MB.

typedef _Float16 v8h  __attribute__((ext_vector_type(8)));
typedef _Float16 v16h __attribute__((ext_vector_type(16)));
typedef float    v8f  __attribute__((ext_vector_type(8)));
typedef int      v4i  __attribute__((vector_size(16)));   // b128 payload type

constexpr int kDModel = 768;
constexpr int kHeads  = 12;
constexpr int kDk     = 64;
constexpr int kBatch  = 2;
constexpr int kSeq    = 2048;
constexpr int kMTot   = kBatch * kSeq;   // 4096
constexpr int kLdsLd  = 48;              // padded LDS leading dim (halves)

#if __has_builtin(__builtin_amdgcn_global_load_async_to_lds_b128)
#define HAVE_ASYNC_LDS 1
typedef __attribute__((address_space(1))) v4i* async_gptr_t;  // global v4i32*
typedef __attribute__((address_space(3))) v4i* async_lptr_t;  // LDS    v4i32*
#else
#define HAVE_ASYNC_LDS 0
#endif

// ---------------------------------------------------------------------------
// Fragment loader: 16x32 f16 tile, row-major, leading dimension `ld` (halves).
// CDNA5 ISA 16-bit A-matrix layout (Bᵀ by symmetry):
//   lanes 0-15  hold row M = lane,    K = {0..7, 16..23}
//   lanes 16-31 hold row M = lane-16, K = {8..15, 24..31}
// ---------------------------------------------------------------------------
static __device__ __forceinline__ v16h load_frag_16x32(const _Float16* __restrict__ base,
                                                       int ld) {
    const int lane = threadIdx.x & 31;
    const int m    = lane & 15;
    const int h    = lane >> 4;
    const _Float16* p = base + (size_t)m * ld + 8 * h;
    v8h lo = *(const v8h*)(p);
    v8h hi = *(const v8h*)(p + 16);
    return __builtin_shufflevector(lo, hi, 0,1,2,3,4,5,6,7,8,9,10,11,12,13,14,15);
}

static __device__ __forceinline__ v8f wmma_f16(v16h a, v16h b, v8f c) {
    // (neg_a, A, neg_b, B, c_mod, C, reuse_a, reuse_b)
    return __builtin_amdgcn_wmma_f32_16x16x32_f16(false, a, false, b, (short)0, c,
                                                  false, false);
}

// Copy 16 consecutive halves (32 B) global -> LDS: async path uses the CDNA5
// GLOBAL_LOAD_ASYNC_TO_LDS_B128 (tracked with ASYNCcnt), fallback is a plain
// load + ds_store pair ordered by the workgroup barrier.
static __device__ __forceinline__ void stage16(_Float16* l, const _Float16* g) {
#if HAVE_ASYNC_LDS
    __builtin_amdgcn_global_load_async_to_lds_b128(
        (async_gptr_t)(_Float16*)g, (async_lptr_t)l, 0, 0);
    __builtin_amdgcn_global_load_async_to_lds_b128(
        (async_gptr_t)(_Float16*)(g + 8), (async_lptr_t)(l + 8), 0, 0);
#else
    *(v8h*)l       = *(const v8h*)g;
    *(v8h*)(l + 8) = *(const v8h*)(g + 8);
#endif
}

static __device__ __forceinline__ void wait_async_copies() {
#if HAVE_ASYNC_LDS
#if __has_builtin(__builtin_amdgcn_s_wait_asynccnt)
    __builtin_amdgcn_s_wait_asynccnt(0);
#else
    asm volatile("s_wait_asynccnt 0x0" ::: "memory");
#endif
#endif
}

static __device__ __forceinline__ float rowmax16(float x) {
    x = fmaxf(x, __shfl_xor(x, 1, 16));
    x = fmaxf(x, __shfl_xor(x, 2, 16));
    x = fmaxf(x, __shfl_xor(x, 4, 16));
    x = fmaxf(x, __shfl_xor(x, 8, 16));
    return x;
}
static __device__ __forceinline__ float rowsum16(float x) {
    x += __shfl_xor(x, 1, 16);
    x += __shfl_xor(x, 2, 16);
    x += __shfl_xor(x, 4, 16);
    x += __shfl_xor(x, 8, 16);
    return x;
}

// ---------------------------------------------------------------------------
// fp32 -> f16 staging pass (bandwidth-bound)
// ---------------------------------------------------------------------------
__global__ void mha_cvt_f32_to_f16(const float* __restrict__ src,
                                   _Float16* __restrict__ dst, int n) {
    int i = blockIdx.x * blockDim.x + threadIdx.x;
    if (i < n) dst[i] = (_Float16)src[i];
}

// ---------------------------------------------------------------------------
// Staged GEMM: y = x @ W^T + bias over (4096 x 768) x (768 x 768).
// Block = 128 threads (4 waves); block tile 64(M) x 64(N); each wave computes
// 16x64 (A fragment reused across 4 WMMAs). Weight k-slab (64x32 f16) is
// double-buffered in LDS, filled with async global->LDS copies.
// mode 0: dsth[b][h][s][d] (Q,K)   mode 1: dsth[b][h][d][s] (V transposed)
// mode 2: dstf = fp32 row-major output (final projection)
// ---------------------------------------------------------------------------
__global__ void __launch_bounds__(128)
mha_gemm_staged(const _Float16* __restrict__ X, const _Float16* __restrict__ W,
                const float* __restrict__ bias, _Float16* __restrict__ dsth,
                float* __restrict__ dstf, int mode) {
    __shared__ _Float16 wtile[2][64 * kLdsLd];

    const int t    = threadIdx.x;
    const int lane = t & 31;
    const int wid  = t >> 5;
    const int col  = lane & 15;
    const int half = lane >> 4;
    const int m0   = blockIdx.x * 64 + wid * 16;
    const int n0   = blockIdx.y * 64;

    // Cooperative staging: thread t copies 16 halves of row (t>>1).
    const int srow = t >> 1;
    const int scol = (t & 1) * 16;
    const _Float16* wsrc  = W + (size_t)(n0 + srow) * kDModel + scol;
    _Float16*       ldst0 = &wtile[0][srow * kLdsLd + scol];
    _Float16*       ldst1 = &wtile[1][srow * kLdsLd + scol];

    const _Float16* xa = X + (size_t)m0 * kDModel;

    stage16(ldst0, wsrc);                       // prologue: k-chunk 0 -> buf 0

    v8f acc0 = {}, acc1 = {}, acc2 = {}, acc3 = {};
    const int kSteps = kDModel / 32;            // 24
    for (int kc = 0; kc < kSteps; ++kc) {
        const int cur = kc & 1;
        wait_async_copies();
        __syncthreads();                        // staged slab visible to block
        if (kc + 1 < kSteps)
            stage16(cur ? ldst0 : ldst1, wsrc + (size_t)(kc + 1) * 32);
        if (kc + 2 < kSteps)
            __builtin_prefetch(xa + (size_t)col * kDModel + (kc + 2) * 32, 0, 1);

        v16h a  = load_frag_16x32(xa + kc * 32, kDModel);
        const _Float16* wb = &wtile[cur][0];
        v16h b0 = load_frag_16x32(wb + 0  * kLdsLd, kLdsLd);
        v16h b1 = load_frag_16x32(wb + 16 * kLdsLd, kLdsLd);
        v16h b2 = load_frag_16x32(wb + 32 * kLdsLd, kLdsLd);
        v16h b3 = load_frag_16x32(wb + 48 * kLdsLd, kLdsLd);
        acc0 = wmma_f16(a, b0, acc0);
        acc1 = wmma_f16(a, b1, acc1);
        acc2 = wmma_f16(a, b2, acc2);
        acc3 = wmma_f16(a, b3, acc3);
    }

    v8f accs[4] = {acc0, acc1, acc2, acc3};
    #pragma unroll
    for (int nt = 0; nt < 4; ++nt) {
        const int   n  = n0 + nt * 16 + col;
        const float bn = bias[n];
        const int   h  = n >> 6;
        const int   d  = n & 63;
        #pragma unroll
        for (int v = 0; v < 8; ++v) {
            int   m = m0 + v + 8 * half;        // D-fragment row layout
            float y = accs[nt][v] + bn;
            if (mode == 2) {
                dstf[(size_t)m * kDModel + n] = y;
            } else {
                int    b_  = m >> 11;           // seq = 2048
                int    s   = m & 2047;
                size_t idx = (mode == 1)
                    ? (((size_t)b_ * kHeads + h) * kDk + d) * kSeq + s
                    : (((size_t)b_ * kHeads + h) * kSeq + s) * kDk + d;
                dsth[idx] = (_Float16)y;
            }
        }
    }
}

// ---------------------------------------------------------------------------
// Flash-attention: one wave per 16-query tile per (b,h). Per 32-key chunk:
// 4 WMMAs for S = Q K^T, online softmax (row stats align per-lane with the
// D-fragment layout M = v + 8*(lane>=16)), P re-layout via wave-private LDS,
// 4 WMMAs for O += P V (V pre-transposed, so Bᵀ tiles are row-major).
// ---------------------------------------------------------------------------
__global__ void __launch_bounds__(32)
mha_attn_wmma(const _Float16* __restrict__ Qh, const _Float16* __restrict__ Kh,
              const _Float16* __restrict__ Vt, const int* __restrict__ mask,
              _Float16* __restrict__ AO) {
    __shared__ _Float16 pbuf[16 * 32];

    const int lane = threadIdx.x & 31;
    const int col  = lane & 15;
    const int half = lane >> 4;
    const int qt   = blockIdx.x;              // query tile 0..127
    const int bh   = blockIdx.y;              // 0..23
    const int b    = bh / kHeads;
    const int hh   = bh % kHeads;

    const _Float16* Qb = Qh + ((size_t)bh * kSeq + (size_t)qt * 16) * kDk;
    const _Float16* Kb = Kh + (size_t)bh * kSeq * kDk;
    const _Float16* Vb = Vt + (size_t)bh * kDk * kSeq;   // [d][s], ld = kSeq
    const int*      mp = mask + b * kSeq;

    v16h q0 = load_frag_16x32(Qb, kDk);
    v16h q1 = load_frag_16x32(Qb + 32, kDk);

    v8f o0 = {}, o1 = {}, o2 = {}, o3 = {};
    float mrow[8], lrow[8];
    #pragma unroll
    for (int v = 0; v < 8; ++v) { mrow[v] = -1.0e30f; lrow[v] = 0.0f; }

    for (int jc = 0; jc < kSeq; jc += 32) {
        v8f s0 = {}, s1 = {};
        {
            v16h ka = load_frag_16x32(Kb + (size_t)jc * kDk, kDk);
            v16h kb = load_frag_16x32(Kb + (size_t)jc * kDk + 32, kDk);
            s0 = wmma_f16(q0, ka, s0);
            s0 = wmma_f16(q1, kb, s0);
        }
        {
            v16h ka = load_frag_16x32(Kb + (size_t)(jc + 16) * kDk, kDk);
            v16h kb = load_frag_16x32(Kb + (size_t)(jc + 16) * kDk + 32, kDk);
            s1 = wmma_f16(q0, ka, s1);
            s1 = wmma_f16(q1, kb, s1);
        }

        const int mk0 = mp[jc + col];
        const int mk1 = mp[jc + 16 + col];
        float p0[8], p1[8], alpha[8];
        #pragma unroll
        for (int v = 0; v < 8; ++v) {
            float a0 = mk0 ? s0[v] * 0.125f : -1.0e9f;   // 1/sqrt(64)
            float a1 = mk1 ? s1[v] * 0.125f : -1.0e9f;
            float rm = rowmax16(fmaxf(a0, a1));
            float mn = fmaxf(mrow[v], rm);
            alpha[v] = __expf(mrow[v] - mn);
            p0[v] = __expf(a0 - mn);
            p1[v] = __expf(a1 - mn);
            lrow[v] = lrow[v] * alpha[v] + rowsum16(p0[v] + p1[v]);
            mrow[v] = mn;
        }
        #pragma unroll
        for (int v = 0; v < 8; ++v) {
            o0[v] *= alpha[v]; o1[v] *= alpha[v];
            o2[v] *= alpha[v]; o3[v] *= alpha[v];
        }

        #pragma unroll
        for (int v = 0; v < 8; ++v) {
            int row = v + 8 * half;
            pbuf[row * 32 + col]      = (_Float16)p0[v];
            pbuf[row * 32 + 16 + col] = (_Float16)p1[v];
        }
        __syncthreads();                       // single-wave WG: near-free
        v16h pa = load_frag_16x32(pbuf, 32);
        __syncthreads();

        o0 = wmma_f16(pa, load_frag_16x32(Vb + jc, kSeq), o0);
        o1 = wmma_f16(pa, load_frag_16x32(Vb + (size_t)16 * kSeq + jc, kSeq), o1);
        o2 = wmma_f16(pa, load_frag_16x32(Vb + (size_t)32 * kSeq + jc, kSeq), o2);
        o3 = wmma_f16(pa, load_frag_16x32(Vb + (size_t)48 * kSeq + jc, kSeq), o3);
    }

    #pragma unroll
    for (int v = 0; v < 8; ++v) {
        float inv = 1.0f / lrow[v];
        int   s   = qt * 16 + v + 8 * half;
        size_t base = ((size_t)b * kSeq + s) * kDModel + (size_t)hh * kDk;
        AO[base + col]      = (_Float16)(o0[v] * inv);
        AO[base + 16 + col] = (_Float16)(o1[v] * inv);
        AO[base + 32 + col] = (_Float16)(o2[v] * inv);
        AO[base + 48 + col] = (_Float16)(o3[v] * inv);
    }
}

// ---------------------------------------------------------------------------
extern "C" void kernel_launch(void* const* d_in, const int* in_sizes, int n_in,
                              void* d_out, int out_size, void* d_ws, size_t ws_size,
                              hipStream_t stream) {
    const float* q    = (const float*)d_in[0];
    const float* k    = (const float*)d_in[1];
    const float* v    = (const float*)d_in[2];
    const int*   mask = (const int*)  d_in[3];
    const float* Wq   = (const float*)d_in[4];
    const float* bq   = (const float*)d_in[5];
    const float* Wk   = (const float*)d_in[6];
    const float* bk   = (const float*)d_in[7];
    const float* Wv   = (const float*)d_in[8];
    const float* bv   = (const float*)d_in[9];
    const float* Wo   = (const float*)d_in[10];
    const float* bo   = (const float*)d_in[11];
    float* out = (float*)d_out;

    const size_t NX = (size_t)kMTot * kDModel;                 // 3,145,728
    const size_t NW = (size_t)kDModel * kDModel;               //   589,824
    const size_t NH = (size_t)kBatch * kHeads * kSeq * kDk;    // == NX

    _Float16* ws   = (_Float16*)d_ws;
    _Float16* xq   = ws;
    _Float16* xk   = xq + NX;
    _Float16* xv   = xk + NX;
    _Float16* wq16 = xv + NX;
    _Float16* wk16 = wq16 + NW;
    _Float16* wv16 = wk16 + NW;
    _Float16* wo16 = wv16 + NW;
    _Float16* Qh   = wo16 + NW;
    _Float16* Kh   = Qh + NH;
    _Float16* Vt   = Kh + NH;
    _Float16* AO   = Vt + NH;                                  // ~48.8 MB total

    const int tb = 256;
    const int gx = (int)((NX + tb - 1) / tb);
    const int gw = (int)((NW + tb - 1) / tb);
    mha_cvt_f32_to_f16<<<gx, tb, 0, stream>>>(q,  xq,   (int)NX);
    mha_cvt_f32_to_f16<<<gx, tb, 0, stream>>>(k,  xk,   (int)NX);
    mha_cvt_f32_to_f16<<<gx, tb, 0, stream>>>(v,  xv,   (int)NX);
    mha_cvt_f32_to_f16<<<gw, tb, 0, stream>>>(Wq, wq16, (int)NW);
    mha_cvt_f32_to_f16<<<gw, tb, 0, stream>>>(Wk, wk16, (int)NW);
    mha_cvt_f32_to_f16<<<gw, tb, 0, stream>>>(Wv, wv16, (int)NW);
    mha_cvt_f32_to_f16<<<gw, tb, 0, stream>>>(Wo, wo16, (int)NW);

    dim3 gg(kMTot / 64, kDModel / 64);       // (64, 12), 4 waves/block
    mha_gemm_staged<<<gg, 128, 0, stream>>>(xq, wq16, bq, Qh, nullptr, 0);
    mha_gemm_staged<<<gg, 128, 0, stream>>>(xk, wk16, bk, Kh, nullptr, 0);
    mha_gemm_staged<<<gg, 128, 0, stream>>>(xv, wv16, bv, Vt, nullptr, 1);

    dim3 ag(kSeq / 16, kBatch * kHeads);     // (128, 24), 1 wave/block
    mha_attn_wmma<<<ag, 32, 0, stream>>>(Qh, Kh, Vt, mask, AO);

    mha_gemm_staged<<<gg, 128, 0, stream>>>(AO, wo16, bo, nullptr, out, 2);

    (void)in_sizes; (void)n_in; (void)out_size; (void)ws_size;
}
